// _ScaledDotProductAttention_39178691674270
// MI455X (gfx1250) — matrix operand order repaired and checked by reading, not compiled, not run on hardware
//
#include <hip/hip_runtime.h>

typedef _Float16 h2  __attribute__((ext_vector_type(2)));
typedef _Float16 h8  __attribute__((ext_vector_type(8)));
typedef _Float16 h16 __attribute__((ext_vector_type(16)));
typedef float    f2  __attribute__((ext_vector_type(2)));
typedef float    f4  __attribute__((ext_vector_type(4)));
typedef float    f8  __attribute__((ext_vector_type(8)));

#define B_ 4
#define H_ 16
#define S_ 1024
#define D_ 64

// ---- helpers -------------------------------------------------------------

__device__ __forceinline__ float wave16_max(float x) {
  x = fmaxf(x, __shfl_xor(x, 1, 32));
  x = fmaxf(x, __shfl_xor(x, 2, 32));
  x = fmaxf(x, __shfl_xor(x, 4, 32));
  x = fmaxf(x, __shfl_xor(x, 8, 32));
  return x;
}
__device__ __forceinline__ float wave16_sum(float x) {
  x += __shfl_xor(x, 1, 32);
  x += __shfl_xor(x, 2, 32);
  x += __shfl_xor(x, 4, 32);
  x += __shfl_xor(x, 8, 32);
  return x;
}
__device__ __forceinline__ h8 ld_h8(const _Float16* p) {
  return *(const h8*)p;   // 16B aligned by construction -> global_load_b128
}
__device__ __forceinline__ h16 cat16(h8 lo, h8 hi) {
  return __builtin_shufflevector(lo, hi, 0,1,2,3,4,5,6,7,8,9,10,11,12,13,14,15);
}

// ---- kernel 1: RoPE(q), RoPE(k^T), v transpose; all to f16 ---------------
// qr, kr: [B*H, S, D] f16 ; vT: [B*H, D, S] f16
__global__ void rope_prep(const float* __restrict__ q, const float* __restrict__ k,
                          const float* __restrict__ v,
                          _Float16* __restrict__ qr, _Float16* __restrict__ kr,
                          _Float16* __restrict__ vT) {
  const size_t idx = (size_t)blockIdx.x * blockDim.x + threadIdx.x;
  const int i  = (int)(idx & 31);          // rotary pair index, D/2 = 32
  const int s  = (int)((idx >> 5) & (S_ - 1));
  const int bh = (int)(idx >> 15);         // / (32*1024)
  if (bh >= B_ * H_) return;

  // theta_i = 10000^(-2i/D);  ln(10000)/32 = 0.2878231366...
  const float ang = (float)s * __expf(-0.2878231366f * (float)i);
  float sn, cs;
  __sincosf(ang, &sn, &cs);

  const size_t rowq = ((size_t)bh * S_ + s) * D_ + 2 * i;
  const size_t kb   = (size_t)bh * D_ * S_;

  // q -> RoPE -> qr (f16); inputs are single-use -> non-temporal loads
  const f2 qp = __builtin_nontemporal_load((const f2*)(q + rowq));
  *(h2*)(qr + rowq) = (h2){(_Float16)(qp.x * cs - qp.y * sn),
                           (_Float16)(qp.y * cs + qp.x * sn)};

  // k is [B,H,D,S]; gather column s, RoPE, store kr row-major [s][d]
  const float k0 = __builtin_nontemporal_load(k + kb + (size_t)(2 * i)     * S_ + s);
  const float k1 = __builtin_nontemporal_load(k + kb + (size_t)(2 * i + 1) * S_ + s);
  *(h2*)(kr + rowq) = (h2){(_Float16)(k0 * cs - k1 * sn),
                           (_Float16)(k1 * cs + k0 * sn)};

  // v -> f16 transposed [d][s] for contiguous WMMA B-fragment loads
  const f2 vp = __builtin_nontemporal_load((const f2*)(v + rowq));
  vT[kb + (size_t)(2 * i)     * S_ + s] = (_Float16)vp.x;
  vT[kb + (size_t)(2 * i + 1) * S_ + s] = (_Float16)vp.y;
}

// ---- fused kernel: scores + softmax + P@V --------------------------------
// One wave per (bh, 16-row q tile). Score panel (16x1024 f32) lives in LDS,
// XOR-swizzled (col ^ 4*row) so phase-1 stores are conflict-free and phase-2
// b128 reads hit every bank exactly twice (the 128-dword/64-bank minimum).
__global__ void __launch_bounds__(32) attn_fused(
    const _Float16* __restrict__ qr, const _Float16* __restrict__ kr,
    const _Float16* __restrict__ vT,
    const float* __restrict__ prev, const float* __restrict__ amask,
    const unsigned char* __restrict__ kpad, const float* __restrict__ scale_p,
    float* __restrict__ scores, float* __restrict__ probs, float* __restrict__ out) {
  extern __shared__ float panel[];   // 16 * 1024 floats = 65536 B
  const int tid  = threadIdx.x;
  const int half = tid >> 4, lm = tid & 15;
  const int qt = blockIdx.x, bh = blockIdx.y, b = bh / H_;
  const int q0 = qt * 16;
  const float scale = scale_p[0];

  // ---------------- phase 1: scores = qr @ kr^T * scale + prev + mask ----
  const _Float16* qrow = qr + ((size_t)bh * S_ + q0 + lm) * D_;
  const h16 a0 = cat16(ld_h8(qrow +      half * 8), ld_h8(qrow + 16 + half * 8));
  const h16 a1 = cat16(ld_h8(qrow + 32 + half * 8), ld_h8(qrow + 48 + half * 8));

  float mrun[8], lrun[8];
#pragma unroll
  for (int j = 0; j < 8; ++j) { mrun[j] = -3.0e38f; lrun[j] = 0.0f; }

  const float* prevrow0 = prev   + ((size_t)bh * S_ + q0) * S_;
  float*       srow0    = scores + ((size_t)bh * S_ + q0) * S_;

  for (int kt = 0; kt < S_ / 16; ++kt) {
    const int col = kt * 16 + lm;
    const _Float16* krow = kr + ((size_t)bh * S_ + col) * D_;
    __builtin_prefetch(krow + 16 * D_, 0, 1);
    const h16 b0 = cat16(ld_h8(krow +      half * 8), ld_h8(krow + 16 + half * 8));
    const h16 b1 = cat16(ld_h8(krow + 32 + half * 8), ld_h8(krow + 48 + half * 8));

    f8 c = {};
    c = __builtin_amdgcn_wmma_f32_16x16x32_f16(false, a0, false, b0, (short)0, c, false, false);
    c = __builtin_amdgcn_wmma_f32_16x16x32_f16(false, a1, false, b1, (short)0, c, false, false);

    const bool pad = kpad[(size_t)b * S_ + col] != 0;
#pragma unroll
    for (int j = 0; j < 8; ++j) {
      const int row = half * 8 + j;               // C layout: M = 8*half + vgpr
      float sc = c[j] * scale
               + __builtin_nontemporal_load(prevrow0 + (size_t)row * S_ + col)
               + amask[(size_t)(q0 + row) * S_ + col];
      if (pad) sc = -1.0e30f;
      __builtin_nontemporal_store(sc, srow0 + (size_t)row * S_ + col);
      panel[row * 1024 + (col ^ (4 * row))] = sc;   // swizzled LDS panel

      // per-lane deferred online softmax stats (this lane's column subset)
      const float mo = mrun[j];
      const float mn = fmaxf(mo, sc);
      lrun[j] = lrun[j] * __expf(mo - mn) + __expf(sc - mn);
      mrun[j] = mn;
    }
  }

  // combine stats across the 16 lanes owning each row
  float mfin[8], lfin[8];
#pragma unroll
  for (int j = 0; j < 8; ++j) {
    mfin[j] = wave16_max(mrun[j]);
    lfin[j] = wave16_sum(lrun[j] * __expf(mrun[j] - mfin[j]));
  }
  // redistribute: phase 2 needs stats of row lm (owned by half-group lm>>3)
  const int src = ((lm >> 3) << 4) | lm;
  float mrow = 0.0f, lrow = 1.0f;
#pragma unroll
  for (int j = 0; j < 8; ++j) {
    const float mj = __shfl(mfin[j], src, 32);
    const float lj = __shfl(lfin[j], src, 32);
    if ((lm & 7) == j) { mrow = mj; lrow = lj; }
  }
  const float invl = 1.0f / lrow;

  __syncthreads();   // single-wave WG: barrier -> s_nop, but fences DS counts

  // ---------------- phase 2: probs = softmax(panel); out = probs @ v -----
  float* prow = probs + ((size_t)bh * S_ + q0 + lm) * S_;
  f8 acc[4];
#pragma unroll
  for (int nt = 0; nt < 4; ++nt) acc[nt] = (f8){};

  const int K = 4 * lm;                 // this lane's row swizzle key
  const float* myrow = panel + lm * 1024;

  for (int kc = 0; kc < S_ / 32; ++kc) {
    const int kb = kc * 32;
    const int c0 = kb + half * 8;       // A elements 0..7
    const int c1 = kb + 16 + half * 8;  // A elements 8..15
    float sv[16];
    *(f4*)(sv + 0)  = *(const f4*)(myrow + ((c0    ) ^ K));
    *(f4*)(sv + 4)  = *(const f4*)(myrow + ((c0 + 4) ^ K));
    *(f4*)(sv + 8)  = *(const f4*)(myrow + ((c1    ) ^ K));
    *(f4*)(sv + 12) = *(const f4*)(myrow + ((c1 + 4) ^ K));

    h16 a;
#pragma unroll
    for (int e = 0; e < 16; ++e) {
      const float p = __expf(sv[e] - mrow) * invl;
      sv[e] = p;
      a[e] = (_Float16)p;
    }
    // write attn_weights (f32), streaming
    __builtin_nontemporal_store(*(const f4*)(sv + 0),  (f4*)(prow + c0));
    __builtin_nontemporal_store(*(const f4*)(sv + 4),  (f4*)(prow + c0 + 4));
    __builtin_nontemporal_store(*(const f4*)(sv + 8),  (f4*)(prow + c1));
    __builtin_nontemporal_store(*(const f4*)(sv + 12), (f4*)(prow + c1 + 4));

#pragma unroll
    for (int nt = 0; nt < 4; ++nt) {
      // B column = output dim d; lane holds vT row d, keys 8h..8h+7 and 16+8h..16+8h+7
      const _Float16* vrow = vT + ((size_t)bh * D_ + nt * 16 + lm) * S_ + kb + half * 8;
      const h16 bf = cat16(ld_h8(vrow), ld_h8(vrow + 16));
      acc[nt] = __builtin_amdgcn_wmma_f32_16x16x32_f16(false, a, false, bf, (short)0,
                                                       acc[nt], false, false);
    }
  }

#pragma unroll
  for (int nt = 0; nt < 4; ++nt)
#pragma unroll
    for (int j = 0; j < 8; ++j) {
      const int row = half * 8 + j;
      __builtin_nontemporal_store(
          acc[nt][j], out + ((size_t)bh * S_ + q0 + row) * D_ + nt * 16 + lm);
    }
}

// ---- launcher ------------------------------------------------------------

extern "C" void kernel_launch(void* const* d_in, const int* in_sizes, int n_in,
                              void* d_out, int out_size, void* d_ws, size_t ws_size,
                              hipStream_t stream) {
  const float* q     = (const float*)d_in[0];   // [B,H,S,D]
  const float* k     = (const float*)d_in[1];   // [B,H,D,S]
  const float* v     = (const float*)d_in[2];   // [B,H,S,D]
  const float* prev  = (const float*)d_in[3];   // [B,H,S,S]
  const float* scale = (const float*)d_in[4];   // scalar
  const float* amask = (const float*)d_in[5];   // [S,S]
  const unsigned char* kpad = (const unsigned char*)d_in[6];  // [B,S] bool

  float* out    = (float*)d_out;                      // [B,H,S,D]
  float* probs  = out   + (size_t)B_ * H_ * S_ * D_;  // [B,H,S,S]
  float* scores = probs + (size_t)B_ * H_ * S_ * S_;  // [B,H,S,S]

  char* ws = (char*)d_ws;
  const size_t nqk = (size_t)B_ * H_ * S_ * D_;
  _Float16* qr = (_Float16*)ws; ws += nqk * 2;
  _Float16* kr = (_Float16*)ws; ws += nqk * 2;
  _Float16* vT = (_Float16*)ws;

  {
    const size_t total = (size_t)B_ * H_ * S_ * (D_ / 2);
    rope_prep<<<dim3((unsigned)((total + 255) / 256)), dim3(256), 0, stream>>>(
        q, k, v, qr, kr, vT);
  }
  dim3 g(S_ / 16, B_ * H_);
  attn_fused<<<g, dim3(32), 64 * 1024, stream>>>(qr, kr, vT, prev, amask, kpad,
                                                 scale, scores, probs, out);
}
